// RNN_944892805797
// MI455X (gfx1250) — matrix operand order repaired and checked by reading, not compiled, not run on hardware
//
#include <hip/hip_runtime.h>

typedef __attribute__((ext_vector_type(16))) _Float16 v16h;
typedef __attribute__((ext_vector_type(8)))  _Float16 v8h;
typedef __attribute__((ext_vector_type(8)))  float    v8f;
typedef __attribute__((ext_vector_type(4)))  float    v4f;

#define HIDDEN 32
#define FF 40
#define SEQ_T 512
#define WAVES_PER_WG 8
#define ROWS_PER_WAVE 16

// W1 path is pre-scaled by -log2(e), so sigmoid(x) = 1 / (1 + 2^x')
__device__ __forceinline__ float sigmoid_from_scaled(float xs) {
    return __builtin_amdgcn_rcpf(1.0f + __builtin_amdgcn_exp2f(xs));
}

__device__ __forceinline__ v8f wmma_f16(v16h a, v16h b, v8f c) {
    // (neg_a, A, neg_b, B, c_mod, C, reuse_a, reuse_b)
    return __builtin_amdgcn_wmma_f32_16x16x32_f16(false, a, false, b, (short)0, c,
                                                  false, false);
}

// Build a 16x32 f16 A-operand from an LDS row: halves 0..15 = row[2*base .. 2*base+15]
// (base = 0 for lanes 0-15, 8 for 16-31)
__device__ __forceinline__ v16h make_a(const _Float16* row, int base) {
    v8h lo = *(const v8h*)(row + 2 * base);
    v8h hi = *(const v8h*)(row + 2 * base + 8);
    return __builtin_shufflevector(lo, hi, 0, 1, 2, 3, 4, 5, 6, 7,
                                   8, 9, 10, 11, 12, 13, 14, 15);
}

__global__ void __launch_bounds__(256, 1)
rnn_scan_wmma(const float* __restrict__ inputs,
              const float* __restrict__ W1hh, const float* __restrict__ b1hh,
              const float* __restrict__ W2hh, const float* __restrict__ b2hh,
              const float* __restrict__ W1ho, const float* __restrict__ b1ho,
              const float* __restrict__ W2ho, const float* __restrict__ b2ho,
              float* __restrict__ out)
{
    __shared__ __align__(16) _Float16 cbuf[WAVES_PER_WG][16][32];  // h state, A layout source
    __shared__ __align__(16) _Float16 sbh [WAVES_PER_WG][16][64];  // sigmoid(hh), K-padded to 64
    __shared__ __align__(16) _Float16 sbo [WAVES_PER_WG][16][64];  // sigmoid(ho), K-padded to 64
    __shared__ __align__(16) float    ubuf[WAVES_PER_WG][16];      // u[t] for the wave's 16 rows

    const int lane    = threadIdx.x & 31;
    const int wave    = threadIdx.x >> 5;
    const int rowbase = (blockIdx.x * WAVES_PER_WG + wave) * ROWS_PER_WAVE;

    const int n16    = lane & 15;          // B/C/D column owned by this lane
    const int half   = lane >> 4;          // 0: lanes 0-15, 1: lanes 16-31
    const int base8  = half * 8;           // A-operand K base / D-row base
    const int koff16 = half * 16;          // B-operand K base

    const float NL2E = -1.44269504088896340736f;  // -log2(e), folded into W1 path

    // ---------------- load weights into f16 B-operand registers -------------
    // GEMM1 weights pre-scaled by -log2(e): WMMA output is already the exp2 arg.
    v16h  B1hhr[3], B1hor[3];
    float w1u_hh[3], w1u_ho[3], b1hh_r[3], b1ho_r[3];
#pragma unroll
    for (int j = 0; j < 3; ++j) {
        const int n = 16 * j + n16;
        const bool valid = (n < FF);
#pragma unroll
        for (int i = 0; i < 16; ++i) {
            B1hhr[j][i] = valid ? (_Float16)(NL2E * W1hh[(koff16 + i) * FF + n]) : (_Float16)0.f;
            B1hor[j][i] = valid ? (_Float16)(NL2E * W1ho[(koff16 + i) * FF + n]) : (_Float16)0.f;
        }
        w1u_hh[j] = valid ? NL2E * W1hh[32 * FF + n] : 0.f;   // u (k=32) row of W1
        w1u_ho[j] = valid ? NL2E * W1ho[32 * FF + n] : 0.f;
        b1hh_r[j] = valid ? NL2E * b1hh[n] : 0.f;
        b1ho_r[j] = valid ? NL2E * b1ho[n] : 0.f;
    }

    v16h  B2hhr[2][2];                     // [k-chunk][n-tile], K padded 40->64
    float b2hh_r[2];
#pragma unroll
    for (int n2 = 0; n2 < 2; ++n2) {
        const int n = 16 * n2 + n16;
#pragma unroll
        for (int c = 0; c < 2; ++c)
#pragma unroll
            for (int i = 0; i < 16; ++i) {
                const int k = 32 * c + koff16 + i;
                B2hhr[c][n2][i] = (k < FF) ? (_Float16)W2hh[k * HIDDEN + n] : (_Float16)0.f;
            }
        b2hh_r[n2] = b2hh[n];
    }

    v16h B2hor[2];                         // N=1 lives in column 0
#pragma unroll
    for (int c = 0; c < 2; ++c)
#pragma unroll
        for (int i = 0; i < 16; ++i) {
            const int k = 32 * c + koff16 + i;
            B2hor[c][i] = (k < FF && n16 == 0) ? (_Float16)W2ho[k] : (_Float16)0.f;
        }
    const float b2ho_r = b2ho[0];

    // ---------------- zero-init per-wave staging (h0 = 0, K-pad = 0) --------
    {
        _Float16* cf = &cbuf[wave][0][0];
        for (int i = lane; i < 16 * 32; i += 32) cf[i] = (_Float16)0.f;
        _Float16* sh = &sbh[wave][0][0];
        _Float16* so = &sbo[wave][0][0];
        for (int i = lane; i < 16 * 64; i += 32) { sh[i] = (_Float16)0.f; so[i] = (_Float16)0.f; }
    }

    const int m_lo = n16;  // A-operand row owned by this lane

    // Software-pipelined input fetch: u(t) lives in a register one step ahead.
    const size_t urow = (size_t)(rowbase + (lane & 15)) * SEQ_T;
    float ucur = 0.f;
    if (lane < 16) ucur = inputs[urow];    // t = 0

    // ---------------- serial scan over time ---------------------------------
#pragma unroll 1
    for (int t = 0; t < SEQ_T; ++t) {
        // Publish u(t) to LDS; immediately issue load of u(t+1). Its loadcnt
        // wait lands at the *next* iteration's ds_store -> full step of slack.
        if (lane < 16) {
            ubuf[wave][lane] = ucur;
            const int tn = (t + 1 < SEQ_T) ? t + 1 : SEQ_T - 1;
            ucur = inputs[urow + tn];
        }
        // Broadcast u[v + base8] for v = 0..7 (matches D-tile row m = v + base8)
        v4f ul = *(const v4f*)&ubuf[wave][base8];
        v4f uh = *(const v4f*)&ubuf[wave][base8 + 4];
        float uv[8];
#pragma unroll
        for (int v = 0; v < 8; ++v) uv[v] = (v < 4) ? ul[v] : uh[v - 4];

        // A operand: h tile (16x32 f16), K = 32 exactly
        v16h Ac = make_a(&cbuf[wave][m_lo][0], base8);

        // ---- GEMM1 hh: C-input = u*w1u' + b1' (scaled); sigmoid -> sbh -----
#pragma unroll
        for (int j = 0; j < 3; ++j) {
            v8f cin;
#pragma unroll
            for (int v = 0; v < 8; ++v) cin[v] = uv[v] * w1u_hh[j] + b1hh_r[j];
            v8f acc = wmma_f16(Ac, B1hhr[j], cin);
            const int n = 16 * j + n16;
#pragma unroll
            for (int v = 0; v < 8; ++v) {
                float s = sigmoid_from_scaled(acc[v]);
                if (n < FF) sbh[wave][v + base8][n] = (_Float16)s;
            }
        }
        // ---- GEMM1 ho: same with W1ho -> sbo -------------------------------
#pragma unroll
        for (int j = 0; j < 3; ++j) {
            v8f cin;
#pragma unroll
            for (int v = 0; v < 8; ++v) cin[v] = uv[v] * w1u_ho[j] + b1ho_r[j];
            v8f acc = wmma_f16(Ac, B1hor[j], cin);
            const int n = 16 * j + n16;
#pragma unroll
            for (int v = 0; v < 8; ++v) {
                float s = sigmoid_from_scaled(acc[v]);
                if (n < FF) sbo[wave][v + base8][n] = (_Float16)s;
            }
        }

        // ---- GEMM2 hh: h_new = s@W2hh + b2hh  (K=64 padded, 2 N-tiles) -----
        {
            v16h As0 = make_a(&sbh[wave][m_lo][0],  base8);
            v16h As1 = make_a(&sbh[wave][m_lo][32], base8);
#pragma unroll
            for (int n2 = 0; n2 < 2; ++n2) {
                v8f acc = {};
                acc = wmma_f16(As0, B2hhr[0][n2], acc);
                acc = wmma_f16(As1, B2hhr[1][n2], acc);
#pragma unroll
                for (int v = 0; v < 8; ++v) {
                    float hval = acc[v] + b2hh_r[n2];
                    cbuf[wave][v + base8][16 * n2 + n16] = (_Float16)hval;
                }
            }
        }

        // ---- GEMM2 ho: y = s@W2ho + b2ho (column 0 of a 16x16 tile) --------
        {
            v16h At0 = make_a(&sbo[wave][m_lo][0],  base8);
            v16h At1 = make_a(&sbo[wave][m_lo][32], base8);
            v8f acc = {};
            acc = wmma_f16(At0, B2hor[0], acc);
            acc = wmma_f16(At1, B2hor[1], acc);
            if (n16 == 0) {
#pragma unroll
                for (int v = 0; v < 8; ++v) {
                    out[(size_t)(rowbase + v + base8) * SEQ_T + t] = acc[v] + b2ho_r;
                }
            }
        }
    }
}

extern "C" void kernel_launch(void* const* d_in, const int* in_sizes, int n_in,
                              void* d_out, int out_size, void* d_ws, size_t ws_size,
                              hipStream_t stream) {
    const float* inputs = (const float*)d_in[0];
    const float* W1hh   = (const float*)d_in[1];
    const float* b1hh   = (const float*)d_in[2];
    const float* W2hh   = (const float*)d_in[3];
    const float* b2hh   = (const float*)d_in[4];
    const float* W1ho   = (const float*)d_in[5];
    const float* b1ho   = (const float*)d_in[6];
    const float* W2ho   = (const float*)d_in[7];
    const float* b2ho   = (const float*)d_in[8];
    float* out = (float*)d_out;

    const int Btot   = in_sizes[0] / SEQ_T;                       // 16384
    const int rowsWG = WAVES_PER_WG * ROWS_PER_WAVE;              // 128
    const int blocks = (Btot + rowsWG - 1) / rowsWG;              // 128

    rnn_scan_wmma<<<blocks, WAVES_PER_WG * 32, 0, stream>>>(
        inputs, W1hh, b1hh, W2hh, b2hh, W1ho, b1ho, W2ho, b2ho, out);
}
